// EncLayer_88536455840236
// MI455X (gfx1250) — matrix operand order repaired and checked by reading, not compiled, not run on hardware
//
#include <hip/hip_runtime.h>
#include <hip/hip_bf16.h>
#include <math.h>

// Problem constants (from reference)
#define BB   4
#define NN   2048
#define KKE  30          // neighbors per node
#define HH   128
#define EEW  128
#define HID  512
#define NODES (BB*NN)    // 8192
#define EPS_LN 1e-5f

typedef __attribute__((ext_vector_type(16))) __bf16 bf16x16;
typedef __attribute__((ext_vector_type(8)))  __bf16 bf16x8;
typedef __attribute__((ext_vector_type(4)))  __bf16 bf16x4;
typedef __attribute__((ext_vector_type(8)))  float  f32x8;
typedef __attribute__((ext_vector_type(4)))  float  f32x4;

union ABFrag { bf16x16 v; bf16x8 h[2]; };

// Branchless erf-GELU (Abramowitz-Stegun 7.1.26, |err| < 1.5e-7):
// one v_rcp_f32 + one v_exp_f32 + ~12 FMAs, no divergence.
__device__ __forceinline__ float gelu_erf(float x) {
  float ax = fabsf(x);
  float t  = __builtin_amdgcn_rcpf(1.0f + 0.3275911f * ax);
  float p  = t * (0.254829592f +
             t * (-0.284496736f +
             t * (1.421413741f +
             t * (-1.453152027f +
             t * 1.061405429f))));
  float e  = __expf(-ax * ax);
  float er = copysignf(1.0f - p * e, x);
  return 0.5f * x * (1.0f + er);
}

__device__ __forceinline__ bf16x4 cvt4(f32x4 v) {
  bf16x4 r;
  r.x = (__bf16)v.x; r.y = (__bf16)v.y; r.z = (__bf16)v.z; r.w = (__bf16)v.w;
  return r;
}

// Two 16x16 output tiles (row blocks 0 and 16 of a 32-row panel) sharing each
// B fragment: per k-chunk -> 1 B load (2x global b128) + 2 A loads + 2 WMMAs.
// aRow0: LDS addr for (lane&15)*stride + ((lane>=16)?8:0); aOff = 16*stride.
template<int KTILES, int NTILES>
__device__ __forceinline__ void wmma_rowpair(const __bf16* aRow0, int aOff,
                                             const __bf16* __restrict__ bsw,
                                             int nt, int lane,
                                             f32x8& acc0, f32x8& acc1) {
#pragma unroll
  for (int kc = 0; kc < KTILES; ++kc) {
    bf16x16 b = *(const bf16x16*)(bsw + ((size_t)(kc * NTILES + nt) * 32 + lane) * 16);
    ABFrag a0, a1;
    a0.h[0] = *(const bf16x8*)(aRow0 + kc * 32);
    a0.h[1] = *(const bf16x8*)(aRow0 + kc * 32 + 16);
    a1.h[0] = *(const bf16x8*)(aRow0 + aOff + kc * 32);
    a1.h[1] = *(const bf16x8*)(aRow0 + aOff + kc * 32 + 16);
    acc0 = __builtin_amdgcn_wmma_f32_16x16x32_bf16(false, a0.v, false, b,
                                                   (short)0, acc0, false, false);
    acc1 = __builtin_amdgcn_wmma_f32_16x16x32_bf16(false, a1.v, false, b,
                                                   (short)0, acc1, false, false);
  }
}

// ---------------------------------------------------------------------------
// Kernel A: convert W (f32, [OUT][IN], y = W x) into WMMA B-fragment-swizzled
// bf16. For tile (kt,nt), lane j holds halves h=0..15:
//   k = kt*32 + (j>=16 ? 8 : 0) + h + (h>=8 ? 8 : 0),  n = nt*16 + (j&15)
// ---------------------------------------------------------------------------
__global__ void swizzle_weights(const float* __restrict__ W,
                                __bf16* __restrict__ dst, int IN, int OUT) {
  int idx = blockIdx.x * blockDim.x + threadIdx.x;   // one per (tile,lane)
  int ntiles = OUT >> 4;
  int total = (IN >> 5) * ntiles * 32;
  if (idx >= total) return;
  int lane = idx & 31;
  int tile = idx >> 5;
  int nt = tile % ntiles;
  int kt = tile / ntiles;
  int o = nt * 16 + (lane & 15);
  int kbase = kt * 32 + ((lane >= 16) ? 8 : 0);
  __bf16* d = dst + (size_t)idx * 16;
#pragma unroll
  for (int h = 0; h < 16; ++h)
    d[h] = (__bf16)W[(size_t)o * IN + (kbase + h + ((h >= 8) ? 8 : 0))];
}

// ---------------------------------------------------------------------------
// Kernel B1: per-node message MLP + K-sum + residual + LN1
// ---------------------------------------------------------------------------
__global__ __launch_bounds__(256)
void node_message_kernel(const float* __restrict__ hV, const float* __restrict__ hE,
                         const int* __restrict__ Eidx, const float* __restrict__ mA,
                         const __bf16* __restrict__ W1sw, const float* __restrict__ W1b,
                         const __bf16* __restrict__ W2sw, const float* __restrict__ W2b,
                         const __bf16* __restrict__ W3sw, const float* __restrict__ W3b,
                         const float* __restrict__ g1, const float* __restrict__ b1,
                         float* __restrict__ hmidf, __bf16* __restrict__ hmidb) {
  __shared__ __align__(16) __bf16 sEV[32 * 392];   // stride 392 halves (conflict-free)
  __shared__ __align__(16) __bf16 sA1[32 * 136];
  __shared__ __align__(16) __bf16 sA2[32 * 136];
  __shared__ float sCol[256];
  __shared__ float sRed[16];

  const int tid  = threadIdx.x;
  const int lane = tid & 31;
  const int wave = tid >> 5;
  const int node = blockIdx.x;                 // b*N + n
  const size_t nodeBase = (size_t)node * HH;
  const size_t edgeBase = (size_t)node * KKE;

  // Build h_EV tile [32 x 384] bf16: [ h_V(self) | h_E | h_V(gathered) ]
  {
    int row = tid >> 3, c8 = tid & 7;
    int gidx = (row < KKE) ? Eidx[edgeBase + row] : 0;
    const float* seg[3];
    seg[0] = hV + nodeBase;
    seg[1] = hE + (edgeBase + row) * (size_t)EEW;
    seg[2] = hV + (size_t)gidx * HH;
    if (row < KKE) {
#pragma unroll
      for (int i = 0; i < 12; ++i) {
        int chunk = i * 8 + c8;            // 0..95
        int col = chunk * 4;               // 0..380
        int sg = chunk >> 5;               // 0,1,2
        f32x4 v = *(const f32x4*)(seg[sg] + (col & 127));
        *(bf16x4*)(sEV + row * 392 + col) = cvt4(v);
      }
    } else {
      bf16x4 z = {};
#pragma unroll
      for (int i = 0; i < 12; ++i) {
        int col = (i * 8 + c8) * 4;
        *(bf16x4*)(sEV + row * 392 + col) = z;
      }
    }
  }
  __syncthreads();

  const int koff  = (lane >= 16) ? 8 : 0;
  const int mlane = lane & 15;
  const int nt    = wave;                      // 8 waves -> 8 N-tiles of 16
  const int Ncol  = nt * 16 + mlane;

  // GEMM1: [32x384] @ W1^T -> gelu -> sA1
  {
    f32x8 a0 = {}, a1 = {};
    wmma_rowpair<12, 8>(sEV + mlane * 392 + koff, 16 * 392, W1sw, nt, lane, a0, a1);
    float bias = W1b[Ncol];
#pragma unroll
    for (int r = 0; r < 8; ++r) {
      sA1[(r + koff) * 136 + Ncol]      = (__bf16)gelu_erf(a0[r] + bias);
      sA1[(16 + r + koff) * 136 + Ncol] = (__bf16)gelu_erf(a1[r] + bias);
    }
  }
  __syncthreads();

  // GEMM2: [32x128] @ W2^T -> gelu -> sA2
  {
    f32x8 a0 = {}, a1 = {};
    wmma_rowpair<4, 8>(sA1 + mlane * 136 + koff, 16 * 136, W2sw, nt, lane, a0, a1);
    float bias = W2b[Ncol];
#pragma unroll
    for (int r = 0; r < 8; ++r) {
      sA2[(r + koff) * 136 + Ncol]      = (__bf16)gelu_erf(a0[r] + bias);
      sA2[(16 + r + koff) * 136 + Ncol] = (__bf16)gelu_erf(a1[r] + bias);
    }
  }
  __syncthreads();

  // GEMM3: [32x128] @ W3^T -> mask_attend -> column sum over K
  {
    f32x8 a0 = {}, a1 = {};
    wmma_rowpair<4, 8>(sA2 + mlane * 136 + koff, 16 * 136, W3sw, nt, lane, a0, a1);
    float bias = W3b[Ncol];
    float psum = 0.f;
#pragma unroll
    for (int r = 0; r < 8; ++r) {
      int M0 = r + koff;
      int M1 = 16 + r + koff;
      psum += (a0[r] + bias) * mA[edgeBase + M0];
      float mk1 = (M1 < KKE) ? mA[edgeBase + M1] : 0.f;
      psum += (a1[r] + bias) * mk1;
    }
    sCol[((lane >= 16) ? 128 : 0) + Ncol] = psum;
  }
  __syncthreads();

  // dh = sum/SCALE ; residual ; LayerNorm over H
  float x = 0.f, s = 0.f, q = 0.f;
  if (tid < HH) {
    float dh = (sCol[tid] + sCol[128 + tid]) * (1.0f / 30.0f);
    x = hV[nodeBase + tid] + dh;
    s = x; q = x * x;
  }
#pragma unroll
  for (int off = 16; off > 0; off >>= 1) {
    s += __shfl_xor(s, off, 32);
    q += __shfl_xor(q, off, 32);
  }
  if (lane == 0) { sRed[wave] = s; sRed[8 + wave] = q; }
  __syncthreads();
  if (tid < HH) {
    float ts = sRed[0] + sRed[1] + sRed[2] + sRed[3];
    float tq = sRed[8] + sRed[9] + sRed[10] + sRed[11];
    float mean = ts * (1.0f / HH);
    float var  = tq * (1.0f / HH) - mean * mean;
    float y = (x - mean) * rsqrtf(var + EPS_LN) * g1[tid] + b1[tid];
    hmidf[nodeBase + tid] = y;
    hmidb[nodeBase + tid] = (__bf16)y;
  }
}

// ---------------------------------------------------------------------------
// Kernel B2: dense FFN (128->512 gelu ->128) + residual + LN2 + mask_V
// One workgroup handles 32 node rows.
// ---------------------------------------------------------------------------
__global__ __launch_bounds__(256)
void ffn_kernel(const float* __restrict__ hmidf, const __bf16* __restrict__ hmidb,
                const __bf16* __restrict__ Winsw, const float* __restrict__ Winb,
                const __bf16* __restrict__ Woutsw, const float* __restrict__ Woutb,
                const float* __restrict__ g2, const float* __restrict__ b2,
                const float* __restrict__ maskV,
                float* __restrict__ outV, __bf16* __restrict__ hnewb) {
  __shared__ __align__(16) __bf16 sX[32 * 136];
  __shared__ __align__(16) __bf16 sH[32 * 520];
  __shared__ __align__(16) float  sY[32 * 132];

  const int tid = threadIdx.x, lane = tid & 31, wave = tid >> 5;
  const int row0 = blockIdx.x * 32;

  { // stage x (already bf16) into LDS
    int row = tid >> 3, c8 = tid & 7;
    const __bf16* src = hmidb + (size_t)(row0 + row) * HH + c8 * 16;
    *(bf16x8*)(sX + row * 136 + c8 * 16)     = *(const bf16x8*)src;
    *(bf16x8*)(sX + row * 136 + c8 * 16 + 8) = *(const bf16x8*)(src + 8);
  }
  __syncthreads();

  const int koff = (lane >= 16) ? 8 : 0;
  const int mlane = lane & 15;

  // GEMM1: [32x128] @ Win^T (ntiles=32) -> gelu -> sH
#pragma unroll
  for (int qq = 0; qq < 4; ++qq) {
    int nt = wave + qq * 8;
    f32x8 a0 = {}, a1 = {};
    wmma_rowpair<4, 32>(sX + mlane * 136 + koff, 16 * 136, Winsw, nt, lane, a0, a1);
    int Ncol = nt * 16 + mlane;
    float bias = Winb[Ncol];
#pragma unroll
    for (int r = 0; r < 8; ++r) {
      sH[(r + koff) * 520 + Ncol]      = (__bf16)gelu_erf(a0[r] + bias);
      sH[(16 + r + koff) * 520 + Ncol] = (__bf16)gelu_erf(a1[r] + bias);
    }
  }
  __syncthreads();

  // GEMM2: [32x512] @ Wout^T -> sY (f32)
  {
    int nt = wave;
    f32x8 a0 = {}, a1 = {};
    wmma_rowpair<16, 8>(sH + mlane * 520 + koff, 16 * 520, Woutsw, nt, lane, a0, a1);
    int Ncol = nt * 16 + mlane;
    float bias = Woutb[Ncol];
#pragma unroll
    for (int r = 0; r < 8; ++r) {
      sY[(r + koff) * 132 + Ncol]      = a0[r] + bias;
      sY[(16 + r + koff) * 132 + Ncol] = a1[r] + bias;
    }
  }
  __syncthreads();

  // residual + LN2 + mask_V ; write final h_V (f32 out + bf16 for edge gather)
  {
    int row = tid >> 3, c8 = tid & 7;
    int gr = row0 + row;
    float xs[16]; float s = 0.f, q2 = 0.f;
#pragma unroll
    for (int i = 0; i < 16; ++i) {
      int c = c8 * 16 + i;
      float v = sY[row * 132 + c] + hmidf[(size_t)gr * HH + c];
      xs[i] = v; s += v; q2 += v * v;
    }
#pragma unroll
    for (int off = 4; off > 0; off >>= 1) {
      s  += __shfl_xor(s, off, 8);
      q2 += __shfl_xor(q2, off, 8);
    }
    float mean = s * (1.0f / HH);
    float var  = q2 * (1.0f / HH) - mean * mean;
    float rs = rsqrtf(var + EPS_LN);
    float mv = maskV[gr];
#pragma unroll
    for (int i = 0; i < 16; ++i) {
      int c = c8 * 16 + i;
      float y = ((xs[i] - mean) * rs * g2[c] + b2[c]) * mv;
      outV[(size_t)gr * HH + c]  = y;
      hnewb[(size_t)gr * HH + c] = (__bf16)y;
    }
  }
}

// ---------------------------------------------------------------------------
// Kernel C: edge-update MLP + residual + per-edge LN over E
// ---------------------------------------------------------------------------
__global__ __launch_bounds__(256)
void edge_update_kernel(const __bf16* __restrict__ hnewb, const float* __restrict__ hE,
                        const int* __restrict__ Eidx,
                        const __bf16* __restrict__ W11sw, const float* __restrict__ W11b,
                        const __bf16* __restrict__ W12sw, const float* __restrict__ W12b,
                        const __bf16* __restrict__ W13sw, const float* __restrict__ W13b,
                        const float* __restrict__ g3, const float* __restrict__ b3,
                        float* __restrict__ outE) {
  __shared__ __align__(16) __bf16 sEV[32 * 392];
  __shared__ __align__(16) __bf16 sA1[32 * 136];
  __shared__ __align__(16) __bf16 sA2[32 * 136];
  __shared__ __align__(16) float  sM[32 * 132];

  const int tid = threadIdx.x, lane = tid & 31, wave = tid >> 5;
  const int node = blockIdx.x;
  const size_t nodeBase = (size_t)node * HH;
  const size_t edgeBase = (size_t)node * KKE;

  // Build h_EV: segments 0 and 2 already bf16 (b64 copies), segment 1 f32->bf16
  {
    int row = tid >> 3, c8 = tid & 7;
    int gidx = (row < KKE) ? Eidx[edgeBase + row] : 0;
    const __bf16* s0 = hnewb + nodeBase;
    const float*  s1 = hE + (edgeBase + row) * (size_t)EEW;
    const __bf16* s2 = hnewb + (size_t)gidx * HH;
    if (row < KKE) {
#pragma unroll
      for (int i = 0; i < 12; ++i) {
        int chunk = i * 8 + c8;
        int col = chunk * 4;
        int sg = chunk >> 5;
        bf16x4 v;
        if (sg == 0)      v = *(const bf16x4*)(s0 + col);
        else if (sg == 1) v = cvt4(*(const f32x4*)(s1 + (col & 127)));
        else              v = *(const bf16x4*)(s2 + (col & 127));
        *(bf16x4*)(sEV + row * 392 + col) = v;
      }
    } else {
      bf16x4 z = {};
#pragma unroll
      for (int i = 0; i < 12; ++i) {
        int col = (i * 8 + c8) * 4;
        *(bf16x4*)(sEV + row * 392 + col) = z;
      }
    }
  }
  __syncthreads();

  const int koff = (lane >= 16) ? 8 : 0;
  const int mlane = lane & 15;
  const int nt = wave;
  const int Ncol = nt * 16 + mlane;

  {
    f32x8 a0 = {}, a1 = {};
    wmma_rowpair<12, 8>(sEV + mlane * 392 + koff, 16 * 392, W11sw, nt, lane, a0, a1);
    float bias = W11b[Ncol];
#pragma unroll
    for (int r = 0; r < 8; ++r) {
      sA1[(r + koff) * 136 + Ncol]      = (__bf16)gelu_erf(a0[r] + bias);
      sA1[(16 + r + koff) * 136 + Ncol] = (__bf16)gelu_erf(a1[r] + bias);
    }
  }
  __syncthreads();

  {
    f32x8 a0 = {}, a1 = {};
    wmma_rowpair<4, 8>(sA1 + mlane * 136 + koff, 16 * 136, W12sw, nt, lane, a0, a1);
    float bias = W12b[Ncol];
#pragma unroll
    for (int r = 0; r < 8; ++r) {
      sA2[(r + koff) * 136 + Ncol]      = (__bf16)gelu_erf(a0[r] + bias);
      sA2[(16 + r + koff) * 136 + Ncol] = (__bf16)gelu_erf(a1[r] + bias);
    }
  }
  __syncthreads();

  {
    f32x8 a0 = {}, a1 = {};
    wmma_rowpair<4, 8>(sA2 + mlane * 136 + koff, 16 * 136, W13sw, nt, lane, a0, a1);
    float bias = W13b[Ncol];
#pragma unroll
    for (int r = 0; r < 8; ++r) {
      sM[(r + koff) * 132 + Ncol]      = a0[r] + bias;
      sM[(16 + r + koff) * 132 + Ncol] = a1[r] + bias;
    }
  }
  __syncthreads();

  // h_E = LN(h_E + m) per edge row
  {
    int row = tid >> 3, c8 = tid & 7;
    if (row < KKE) {
      const float* eRow = hE + (edgeBase + row) * (size_t)EEW;
      float xs[16]; float s = 0.f, q2 = 0.f;
#pragma unroll
      for (int i = 0; i < 16; ++i) {
        int c = c8 * 16 + i;
        float v = eRow[c] + sM[row * 132 + c];
        xs[i] = v; s += v; q2 += v * v;
      }
#pragma unroll
      for (int off = 4; off > 0; off >>= 1) {
        s  += __shfl_xor(s, off, 8);
        q2 += __shfl_xor(q2, off, 8);
      }
      float mean = s * (1.0f / EEW);
      float var  = q2 * (1.0f / EEW) - mean * mean;
      float rs = rsqrtf(var + EPS_LN);
#pragma unroll
      for (int i = 0; i < 16; ++i) {
        int c = c8 * 16 + i;
        outE[(edgeBase + row) * (size_t)EEW + c] =
            (xs[i] - mean) * rs * g3[c] + b3[c];
      }
    }
  }
}

// ---------------------------------------------------------------------------
extern "C" void kernel_launch(void* const* d_in, const int* in_sizes, int n_in,
                              void* d_out, int out_size, void* d_ws, size_t ws_size,
                              hipStream_t stream) {
  const float* hV    = (const float*)d_in[0];
  const float* hE    = (const float*)d_in[1];
  const int*   Eidx  = (const int*)d_in[2];
  const float* maskV = (const float*)d_in[3];
  const float* maskA = (const float*)d_in[4];
  const float* W1w = (const float*)d_in[5];   const float* W1b = (const float*)d_in[6];
  const float* W2w = (const float*)d_in[7];   const float* W2b = (const float*)d_in[8];
  const float* W3w = (const float*)d_in[9];   const float* W3b = (const float*)d_in[10];
  const float* W11w = (const float*)d_in[11]; const float* W11b = (const float*)d_in[12];
  const float* W12w = (const float*)d_in[13]; const float* W12b = (const float*)d_in[14];
  const float* W13w = (const float*)d_in[15]; const float* W13b = (const float*)d_in[16];
  const float* Winw = (const float*)d_in[17]; const float* Winb = (const float*)d_in[18];
  const float* Woutw = (const float*)d_in[19]; const float* Woutb = (const float*)d_in[20];
  const float* g1 = (const float*)d_in[21]; const float* b1 = (const float*)d_in[22];
  const float* g2 = (const float*)d_in[23]; const float* b2 = (const float*)d_in[24];
  const float* g3 = (const float*)d_in[25]; const float* b3 = (const float*)d_in[26];

  char* ws = (char*)d_ws;
  // bf16 swizzled weights (first ~0.6 MB of ws)
  __bf16* W1sw  = (__bf16*)ws;
  __bf16* W2sw  = W1sw  + 49152;   // 384*128
  __bf16* W3sw  = W2sw  + 16384;   // 128*128
  __bf16* W11sw = W3sw  + 16384;
  __bf16* W12sw = W11sw + 49152;
  __bf16* W13sw = W12sw + 16384;
  __bf16* Winsw = W13sw + 16384;   // 128*512
  __bf16* Woutsw = Winsw + 65536;  // 512*128
  // activations
  float*  hmidf = (float*)(ws + (1u << 20));           // 4 MB  [8192][128] f32
  __bf16* hmidb = (__bf16*)(ws + (5u << 20));          // 2 MB  bf16
  __bf16* hnewb = (__bf16*)(ws + (7u << 20));          // 2 MB  bf16 (final h_V)

  float* outV = (float*)d_out;
  float* outE = outV + (size_t)NODES * HH;

  // --- weight swizzle (tiny) ---
  swizzle_weights<<<(12 * 8 * 32 + 255) / 256, 256, 0, stream>>>(W1w,  W1sw,  384, 128);
  swizzle_weights<<<(4 * 8 * 32 + 255) / 256, 256, 0, stream>>>(W2w,  W2sw,  128, 128);
  swizzle_weights<<<(4 * 8 * 32 + 255) / 256, 256, 0, stream>>>(W3w,  W3sw,  128, 128);
  swizzle_weights<<<(12 * 8 * 32 + 255) / 256, 256, 0, stream>>>(W11w, W11sw, 384, 128);
  swizzle_weights<<<(4 * 8 * 32 + 255) / 256, 256, 0, stream>>>(W12w, W12sw, 128, 128);
  swizzle_weights<<<(4 * 8 * 32 + 255) / 256, 256, 0, stream>>>(W13w, W13sw, 128, 128);
  swizzle_weights<<<(4 * 32 * 32 + 255) / 256, 256, 0, stream>>>(Winw, Winsw, 128, 512);
  swizzle_weights<<<(16 * 8 * 32 + 255) / 256, 256, 0, stream>>>(Woutw, Woutsw, 512, 128);

  // --- node message + LN1 ---
  node_message_kernel<<<NODES, 256, 0, stream>>>(hV, hE, Eidx, maskA,
                                                 W1sw, W1b, W2sw, W2b, W3sw, W3b,
                                                 g1, b1, hmidf, hmidb);
  // --- FFN + LN2 + mask_V ---
  ffn_kernel<<<NODES / 32, 256, 0, stream>>>(hmidf, hmidb, Winsw, Winb, Woutsw, Woutb,
                                             g2, b2, maskV, outV, hnewb);
  // --- edge update + LN3 ---
  edge_update_kernel<<<NODES, 256, 0, stream>>>(hnewb, hE, Eidx,
                                                W11sw, W11b, W12sw, W12b, W13sw, W13b,
                                                g3, b3, outE);
}
